// OptimalPhasorModel_9251359556071
// MI455X (gfx1250) — matrix operand order repaired and checked by reading, not compiled, not run on hardware
//
#include <hip/hip_runtime.h>

// ---------------------------------------------------------------------------
// MI455X (gfx1250) implementation of the phasor model.
// Dominant cost: [8192,1024]x[1024,32000] head GEMM (~537 GFLOP) -> bf16 WMMA
// (v_wmma_f32_16x16x32_bf16, f32 accumulate) with double-buffered LDS staging
// via GLOBAL_LOAD_ASYNC_TO_LDS (ASYNCcnt) when the toolchain exposes it.
// ---------------------------------------------------------------------------

typedef __attribute__((ext_vector_type(16))) __bf16       v16bf;
typedef __attribute__((ext_vector_type(8)))  __bf16       v8bf;
typedef __attribute__((ext_vector_type(8)))  float        v8f;
typedef __attribute__((ext_vector_type(4)))  float        f32x4;
typedef __attribute__((ext_vector_type(4)))  unsigned int u32x4;
typedef __attribute__((ext_vector_type(2)))  unsigned int u32x2;
typedef int i32v4 __attribute__((vector_size(16)));   // matches async-LDS builtin arg

#define Bsz 4
#define Lsz 2048
#define Dsz 1024
#define Vsz 32000
#define MROWS (Bsz * Lsz)      // 8192
#define NCHUNK 16
#define CLEN   128             // NCHUNK*CLEN == Lsz

#if defined(__gfx1250__) && __has_builtin(__builtin_amdgcn_global_load_async_to_lds_b128)
#define USE_ASYNC_LDS 1
#else
#define USE_ASYNC_LDS 0
#endif

#define AS1 __attribute__((address_space(1)))
#define AS3 __attribute__((address_space(3)))

__device__ __forceinline__ void wait_async_zero() {
#if defined(__gfx1250__)
#if __has_builtin(__builtin_amdgcn_s_wait_asynccnt)
  __builtin_amdgcn_s_wait_asynccnt(0);
#else
  asm volatile("s_wait_asynccnt 0x0" ::: "memory");
#endif
#endif
}

__device__ __forceinline__ unsigned short f32_to_bf16(float f) {
  unsigned int u = __float_as_uint(f);
  u = (u + 0x7FFFu + ((u >> 16) & 1u)) >> 16;
  return (unsigned short)u;
}
__device__ __forceinline__ unsigned int bf16pack2(float a, float b) {
  return (unsigned int)f32_to_bf16(a) | ((unsigned int)f32_to_bf16(b) << 16);
}

// ---------------------------------------------------------------------------
// Weight cast + transpose: Wt[n,k] = bf16(W[k,n]).  32x32 LDS tiles.
// ---------------------------------------------------------------------------
__global__ __launch_bounds__(256)
void cast_transpose_kernel(const float* __restrict__ W,
                           unsigned short* __restrict__ Wt, int K, int N) {
  __shared__ float tile[32][33];
  const int n0 = blockIdx.x * 32, k0 = blockIdx.y * 32;
  const int tx = threadIdx.x, ty = threadIdx.y;          // (32, 8)
#pragma unroll
  for (int j = 0; j < 4; ++j) {
    int k = ty + j * 8;
    tile[k][tx] = W[(size_t)(k0 + k) * N + (n0 + tx)];
  }
  __syncthreads();
#pragma unroll
  for (int j = 0; j < 4; ++j) {
    int n = ty + j * 8;
    Wt[(size_t)(n0 + n) * K + (k0 + tx)] = f32_to_bf16(tile[tx][n]);
  }
}

// ---------------------------------------------------------------------------
// Embedding gather: one block per row, b128 copies.
// ---------------------------------------------------------------------------
__global__ __launch_bounds__(256)
void gather_kernel(const int* __restrict__ tokens, const float* __restrict__ embed,
                   float* __restrict__ h) {
  const int row = blockIdx.x;
  const int tok = tokens[row];
  const f32x4* src = (const f32x4*)(embed + (size_t)tok * Dsz);
  f32x4* dst = (f32x4*)(h + (size_t)row * Dsz);
  dst[threadIdx.x] = src[threadIdx.x];
}

// ---------------------------------------------------------------------------
// LayerNorm over D=1024; optional f32 and bf16 outputs.
// ---------------------------------------------------------------------------
__global__ __launch_bounds__(256)
void layernorm_kernel(const float* __restrict__ x, const float* __restrict__ g,
                      const float* __restrict__ b, float* __restrict__ of,
                      unsigned short* __restrict__ obf) {
  __shared__ float red[256], red2[256];
  const int row = blockIdx.x, t = threadIdx.x;
  const f32x4 xv = ((const f32x4*)(x + (size_t)row * Dsz))[t];
  float s  = xv[0] + xv[1] + xv[2] + xv[3];
  float ss = xv[0]*xv[0] + xv[1]*xv[1] + xv[2]*xv[2] + xv[3]*xv[3];
  red[t] = s; red2[t] = ss;
  __syncthreads();
  for (int off = 128; off > 0; off >>= 1) {
    if (t < off) { red[t] += red[t + off]; red2[t] += red2[t + off]; }
    __syncthreads();
  }
  const float mu  = red[0] * (1.0f / Dsz);
  const float var = red2[0] * (1.0f / Dsz) - mu * mu;
  const float rs  = rsqrtf(var + 1e-5f);
  const f32x4 gv = ((const f32x4*)g)[t];
  const f32x4 bv = ((const f32x4*)b)[t];
  f32x4 y;
#pragma unroll
  for (int i = 0; i < 4; ++i) y[i] = (xv[i] - mu) * rs * gv[i] + bv[i];
  if (of)  ((f32x4*)(of + (size_t)row * Dsz))[t] = y;
  if (obf) {
    u32x2 p; p[0] = bf16pack2(y[0], y[1]); p[1] = bf16pack2(y[2], y[3]);
    ((u32x2*)(obf + (size_t)row * Dsz))[t] = p;
  }
}

// ---------------------------------------------------------------------------
// Phasor cumulative memory, two-pass chunked scan over L.
// ---------------------------------------------------------------------------
__global__ __launch_bounds__(256)
void scan_partial_kernel(const float* __restrict__ v, const float* __restrict__ ph,
                         float* __restrict__ pr, float* __restrict__ pi) {
  const int d  = blockIdx.y * 256 + threadIdx.x;
  const int bb = blockIdx.x / NCHUNK;
  const int c  = blockIdx.x % NCHUNK;
  float sr = 0.f, si = 0.f;
  for (int i = 0; i < CLEN; ++i) {
    const int l = c * CLEN + i;
    const float p  = ph[(size_t)l * Dsz + d];
    const float vv = v[((size_t)bb * Lsz + l) * Dsz + d];
    sr += vv * cosf(p);
    si += vv * sinf(p);
  }
  pr[((size_t)bb * NCHUNK + c) * Dsz + d] = sr;
  pi[((size_t)bb * NCHUNK + c) * Dsz + d] = si;
}

__global__ __launch_bounds__(256)
void scan_apply_kernel(const float* __restrict__ ph, const float* __restrict__ pr,
                       const float* __restrict__ pi, float* __restrict__ v,
                       float scale) {
  const int d  = blockIdx.y * 256 + threadIdx.x;
  const int bb = blockIdx.x / NCHUNK;
  const int c  = blockIdx.x % NCHUNK;
  float rr = 0.f, ri = 0.f;
  for (int c2 = 0; c2 < c; ++c2) {
    rr += pr[((size_t)bb * NCHUNK + c2) * Dsz + d];
    ri += pi[((size_t)bb * NCHUNK + c2) * Dsz + d];
  }
  for (int i = 0; i < CLEN; ++i) {
    const int l = c * CLEN + i;
    const float p  = ph[(size_t)l * Dsz + d];
    const float cs = cosf(p), sn = sinf(p);
    const size_t idx = ((size_t)bb * Lsz + l) * Dsz + d;
    const float vv = v[idx];
    rr += vv * cs;
    ri += vv * sn;
    v[idx] = (rr * cs + ri * sn) * scale;   // in-place: retrieved overwrites v
  }
}

// ---------------------------------------------------------------------------
// bf16 WMMA GEMM: C[M,N] = A[M,K] * Bt[N,K]^T + bias (+res0 +res1).
// Block tile 128x128x64, 8 waves, each wave 32(M)x64(N) = 2x4 fragments.
// Double-buffered LDS; staging via async global->LDS DMA when available.
// ---------------------------------------------------------------------------
#define BM 128
#define BN 128
#define BK 64
#define LDT 72   // LDS row stride in bf16 (64 + 8 pad, keeps 16B alignment)

__global__ __launch_bounds__(256)
void gemm_bf16_kernel(const unsigned short* __restrict__ A,
                      const unsigned short* __restrict__ Bt,
                      const float* __restrict__ bias,
                      const float* __restrict__ res0,
                      const float* __restrict__ res1,
                      float* __restrict__ C, int M, int N, int K) {
  __shared__ __align__(16) unsigned short As[2][BM * LDT];
  __shared__ __align__(16) unsigned short Bs[2][BN * LDT];

  const int tid  = threadIdx.x;
  const int lane = tid & 31;
  const int wave = tid >> 5;
  const int wm   = wave & 3;      // M position (x32)
  const int wn   = wave >> 2;     // N position (x64)
  const int m0   = blockIdx.y * BM;
  const int n0   = blockIdx.x * BN;
  const int lrow = tid >> 1;             // 0..127 loader row
  const int lq   = (tid & 1) * 4;        // which half of the 128B row

  const unsigned short* gArow = A  + (size_t)(m0 + lrow) * K;
  const unsigned short* gBrow = Bt + (size_t)(n0 + lrow) * K;

  v8f acc[2][4];
  const v8f vzero = {};
#pragma unroll
  for (int mi = 0; mi < 2; ++mi)
#pragma unroll
    for (int ni = 0; ni < 4; ++ni) acc[mi][ni] = vzero;

  // -------- one K-tile of WMMA work from LDS buffer `buf` --------
  auto compute = [&](int buf) {
#pragma unroll
    for (int kk = 0; kk < BK; kk += 32) {
      // A fragment (16x32 bf16): lane m=lane&15; K runs [kb,kb+8) and [kb+16,kb+24)
      const int kbA = kk + ((lane >> 4) << 3);
      // B fragment (32x16 bf16): lane n=lane&15; K runs [kb,kb+8) and [kb+16,kb+24)
      const int kbB = kk + ((lane >> 4) << 3);
      v16bf af[2], bfm[4];
#pragma unroll
      for (int mi = 0; mi < 2; ++mi) {
        const int r = wm * 32 + mi * 16 + (lane & 15);
        const v8bf lo = *(const v8bf*)&As[buf][r * LDT + kbA];
        const v8bf hi = *(const v8bf*)&As[buf][r * LDT + kbA + 16];
        af[mi] = __builtin_shufflevector(lo, hi, 0,1,2,3,4,5,6,7,8,9,10,11,12,13,14,15);
      }
#pragma unroll
      for (int ni = 0; ni < 4; ++ni) {
        const int r = wn * 64 + ni * 16 + (lane & 15);
        const v8bf lo = *(const v8bf*)&Bs[buf][r * LDT + kbB];
        const v8bf hi = *(const v8bf*)&Bs[buf][r * LDT + kbB + 16];
        bfm[ni] = __builtin_shufflevector(lo, hi, 0,1,2,3,4,5,6,7,8,9,10,11,12,13,14,15);
      }
#pragma unroll
      for (int mi = 0; mi < 2; ++mi)
#pragma unroll
        for (int ni = 0; ni < 4; ++ni)
          acc[mi][ni] = __builtin_amdgcn_wmma_f32_16x16x32_bf16(
              false, af[mi], false, bfm[ni], (short)0, acc[mi][ni], false, false);
    }
  };

#if USE_ASYNC_LDS
  // -------- async DMA staging: global -> LDS, tracked by ASYNCcnt --------
  auto stage_async = [&](int buf, int k0) {
#pragma unroll
    for (int j = 0; j < 4; ++j) {
      __builtin_amdgcn_global_load_async_to_lds_b128(
          (AS1 i32v4*)(gArow + k0 + (size_t)(lq + j) * 8),
          (AS3 i32v4*)&As[buf][lrow * LDT + (lq + j) * 8], 0, 0);
      __builtin_amdgcn_global_load_async_to_lds_b128(
          (AS1 i32v4*)(gBrow + k0 + (size_t)(lq + j) * 8),
          (AS3 i32v4*)&Bs[buf][lrow * LDT + (lq + j) * 8], 0, 0);
    }
  };

  stage_async(0, 0);
  wait_async_zero();
  __syncthreads();
  int cur = 0;
  for (int k0 = 0; k0 < K; k0 += BK) {
    const bool has_next = (k0 + BK) < K;
    if (has_next) stage_async(cur ^ 1, k0 + BK);       // DMA next tile
    if (k0 + 2 * BK < K) {                             // L2 prefetch 2 tiles ahead
      __builtin_prefetch(gArow + k0 + 2 * BK, 0, 1);
      __builtin_prefetch(gBrow + k0 + 2 * BK, 0, 1);
    }
    compute(cur);                                      // WMMAs overlap the DMA
    wait_async_zero();
    __syncthreads();
    cur ^= 1;
  }
#else
  // -------- register double-buffered staging fallback --------
  {
    const u32x4* gA = (const u32x4*)(gArow);
    const u32x4* gB = (const u32x4*)(gBrow);
#pragma unroll
    for (int j = 0; j < 4; ++j) {
      *(u32x4*)&As[0][lrow * LDT + (lq + j) * 8] = gA[lq + j];
      *(u32x4*)&Bs[0][lrow * LDT + (lq + j) * 8] = gB[lq + j];
    }
  }
  __syncthreads();
  int cur = 0;
  for (int k0 = 0; k0 < K; k0 += BK) {
    const bool has_next = (k0 + BK) < K;
    u32x4 ra[4], rb[4];
    if (has_next) {                                    // issue loads early
      const u32x4* gA = (const u32x4*)(gArow + k0 + BK);
      const u32x4* gB = (const u32x4*)(gBrow + k0 + BK);
#pragma unroll
      for (int j = 0; j < 4; ++j) { ra[j] = gA[lq + j]; rb[j] = gB[lq + j]; }
    }
    if (k0 + 2 * BK < K) {
      __builtin_prefetch(gArow + k0 + 2 * BK, 0, 1);
      __builtin_prefetch(gBrow + k0 + 2 * BK, 0, 1);
    }
    compute(cur);                                      // hide load latency
    if (has_next) {
#pragma unroll
      for (int j = 0; j < 4; ++j) {
        *(u32x4*)&As[cur ^ 1][lrow * LDT + (lq + j) * 8] = ra[j];
        *(u32x4*)&Bs[cur ^ 1][lrow * LDT + (lq + j) * 8] = rb[j];
      }
    }
    __syncthreads();
    cur ^= 1;
  }
#endif

  // ---- epilogue: C = acc + bias[n] (+res0 +res1) ----
#pragma unroll
  for (int mi = 0; mi < 2; ++mi) {
#pragma unroll
    for (int ni = 0; ni < 4; ++ni) {
      const int col  = n0 + wn * 64 + ni * 16 + (lane & 15);
      const int rowb = m0 + wm * 32 + mi * 16 + ((lane >> 4) << 3);
      const float bv = bias[col];
#pragma unroll
      for (int r = 0; r < 8; ++r) {
        const size_t idx = (size_t)(rowb + r) * N + col;
        float o = acc[mi][ni][r] + bv;
        if (res0) o += res0[idx];
        if (res1) o += res1[idx];
        C[idx] = o;
      }
    }
  }
}

// ---------------------------------------------------------------------------
// Host-side orchestration
// ---------------------------------------------------------------------------
extern "C" void kernel_launch(void* const* d_in, const int* in_sizes, int n_in,
                              void* d_out, int out_size, void* d_ws, size_t ws_size,
                              hipStream_t stream) {
  (void)in_sizes; (void)n_in; (void)out_size; (void)ws_size;
  const int*   tokens = (const int*)  d_in[0];
  const float* embed  = (const float*)d_in[1];
  const float* ln1_g  = (const float*)d_in[2];
  const float* ln1_b  = (const float*)d_in[3];
  const float* ph1    = (const float*)d_in[4];
  const float* wv1    = (const float*)d_in[5];
  const float* bv1    = (const float*)d_in[6];
  const float* lno1_g = (const float*)d_in[7];
  const float* lno1_b = (const float*)d_in[8];
  const float* wo1    = (const float*)d_in[9];
  const float* bo1    = (const float*)d_in[10];
  const float* ln2_g  = (const float*)d_in[11];
  const float* ln2_b  = (const float*)d_in[12];
  const float* ph2    = (const float*)d_in[13];
  const float* wv2    = (const float*)d_in[14];
  const float* bv2    = (const float*)d_in[15];
  const float* lno2_g = (const float*)d_in[16];
  const float* lno2_b = (const float*)d_in[17];
  const float* wo2    = (const float*)d_in[18];
  const float* bo2    = (const float*)d_in[19];
  const float* ln3_g  = (const float*)d_in[20];
  const float* ln3_b  = (const float*)d_in[21];
  const float* w_head = (const float*)d_in[22];
  const float* b_head = (const float*)d_in[23];
  float* out = (float*)d_out;

  // Workspace carving (~192 MB)
  char* ws = (char*)d_ws;
  float* h    = (float*)ws;          ws += (size_t)MROWS * Dsz * 4;
  float* xln  = (float*)ws;          ws += (size_t)MROWS * Dsz * 4;
  unsigned short* xbf = (unsigned short*)ws; ws += (size_t)MROWS * Dsz * 2;
  float* v    = (float*)ws;          ws += (size_t)MROWS * Dsz * 4;
  float* pr   = (float*)ws;          ws += (size_t)Bsz * NCHUNK * Dsz * 4;
  float* pim  = (float*)ws;          ws += (size_t)Bsz * NCHUNK * Dsz * 4;
  unsigned short* wv1t = (unsigned short*)ws; ws += (size_t)Dsz * Dsz * 2;
  unsigned short* wo1t = (unsigned short*)ws; ws += (size_t)Dsz * Dsz * 2;
  unsigned short* wv2t = (unsigned short*)ws; ws += (size_t)Dsz * Dsz * 2;
  unsigned short* wo2t = (unsigned short*)ws; ws += (size_t)Dsz * Dsz * 2;
  unsigned short* wht  = (unsigned short*)ws; ws += (size_t)Vsz * Dsz * 2;

  const dim3 blkT(32, 8);
  const dim3 grdDD(Dsz / 32, Dsz / 32);
  hipLaunchKernelGGL(cast_transpose_kernel, grdDD, blkT, 0, stream, wv1, wv1t, Dsz, Dsz);
  hipLaunchKernelGGL(cast_transpose_kernel, grdDD, blkT, 0, stream, wo1, wo1t, Dsz, Dsz);
  hipLaunchKernelGGL(cast_transpose_kernel, grdDD, blkT, 0, stream, wv2, wv2t, Dsz, Dsz);
  hipLaunchKernelGGL(cast_transpose_kernel, grdDD, blkT, 0, stream, wo2, wo2t, Dsz, Dsz);
  hipLaunchKernelGGL(cast_transpose_kernel, dim3(Vsz / 32, Dsz / 32), blkT, 0, stream,
                     w_head, wht, Dsz, Vsz);

  hipLaunchKernelGGL(gather_kernel, dim3(MROWS), dim3(256), 0, stream, tokens, embed, h);

  const dim3 gemmDD(Dsz / BN, MROWS / BM);       // (8, 64)
  const dim3 scanG(Bsz * NCHUNK, Dsz / 256);     // (64, 4)
  const float scale = 0.03125f;                  // 1/sqrt(1024)

  // ---- layer 1 ----
  hipLaunchKernelGGL(layernorm_kernel, dim3(MROWS), dim3(256), 0, stream, h, ln1_g, ln1_b, xln, xbf);
  hipLaunchKernelGGL(gemm_bf16_kernel, gemmDD, dim3(256), 0, stream,
                     xbf, wv1t, bv1, (const float*)nullptr, (const float*)nullptr, v,
                     MROWS, Dsz, Dsz);
  hipLaunchKernelGGL(scan_partial_kernel, scanG, dim3(256), 0, stream, v, ph1, pr, pim);
  hipLaunchKernelGGL(scan_apply_kernel,  scanG, dim3(256), 0, stream, ph1, pr, pim, v, scale);
  hipLaunchKernelGGL(layernorm_kernel, dim3(MROWS), dim3(256), 0, stream, v, lno1_g, lno1_b,
                     (float*)nullptr, xbf);
  hipLaunchKernelGGL(gemm_bf16_kernel, gemmDD, dim3(256), 0, stream,
                     xbf, wo1t, bo1, (const float*)h, (const float*)xln, h, MROWS, Dsz, Dsz);

  // ---- layer 2 ----
  hipLaunchKernelGGL(layernorm_kernel, dim3(MROWS), dim3(256), 0, stream, h, ln2_g, ln2_b, xln, xbf);
  hipLaunchKernelGGL(gemm_bf16_kernel, gemmDD, dim3(256), 0, stream,
                     xbf, wv2t, bv2, (const float*)nullptr, (const float*)nullptr, v,
                     MROWS, Dsz, Dsz);
  hipLaunchKernelGGL(scan_partial_kernel, scanG, dim3(256), 0, stream, v, ph2, pr, pim);
  hipLaunchKernelGGL(scan_apply_kernel,  scanG, dim3(256), 0, stream, ph2, pr, pim, v, scale);
  hipLaunchKernelGGL(layernorm_kernel, dim3(MROWS), dim3(256), 0, stream, v, lno2_g, lno2_b,
                     (float*)nullptr, xbf);
  hipLaunchKernelGGL(gemm_bf16_kernel, gemmDD, dim3(256), 0, stream,
                     xbf, wo2t, bo2, (const float*)h, (const float*)xln, h, MROWS, Dsz, Dsz);

  // ---- head ----
  hipLaunchKernelGGL(layernorm_kernel, dim3(MROWS), dim3(256), 0, stream, h, ln3_g, ln3_b,
                     (float*)nullptr, xbf);
  hipLaunchKernelGGL(gemm_bf16_kernel, dim3(Vsz / BN, MROWS / BM), dim3(256), 0, stream,
                     xbf, wht, b_head, (const float*)nullptr, (const float*)nullptr, out,
                     MROWS, Vsz, Dsz);
}
